// DeformableTransformerDecoderLayer_4861902979349
// MI455X (gfx1250) — matrix operand order repaired and checked by reading, not compiled, not run on hardware
//
#include <hip/hip_runtime.h>

// ---------------------------------------------------------------------------
// DeformableTransformerDecoderLayer for MI455X (gfx1250), wave32 + WMMA f16.
// GEMMs run on v_wmma_f32_16x16x32_f16. Where legal (K%32==0, no input
// activation), tiles are staged with GLOBAL_LOAD_ASYNC_TO_LDS_B128 into a
// ping-pong f32 LDS buffer synchronized with s_wait_asynccnt; otherwise a
// branch-free register-double-buffered f16 staging path is used.
// ---------------------------------------------------------------------------

typedef __attribute__((ext_vector_type(16))) _Float16 v16h;
typedef __attribute__((ext_vector_type(8)))  _Float16 v8h;
typedef __attribute__((ext_vector_type(4)))  _Float16 v4h;
typedef __attribute__((ext_vector_type(8)))  float    v8f;
typedef __attribute__((ext_vector_type(4)))  float    v4f;
typedef __attribute__((__vector_size__(16))) int      i32x4;   // builtin's pointee type

#if defined(__has_builtin)
#if __has_builtin(__builtin_amdgcn_global_load_async_to_lds_b128) && \
    __has_builtin(__builtin_amdgcn_s_wait_asynccnt)
#define HAVE_ASYNC 1
#endif
#endif
#ifndef HAVE_ASYNC
#define HAVE_ASYNC 0
#endif

// ---------------------------- generic WMMA GEMM ----------------------------
// Y[m,n] = act_out( scale * sum_k act_in(X[m,k]) * B[k,n] + bias[n] + res[m,n] )
// B[k,n] = W[n*ldb + k]   (BTRANS==0, weight stored (N,K) row-major)
// B[k,n] = W[k*ldb + n]   (BTRANS==1, e.g. attention P @ V)
// Two-level batch: z = bi*Hdiv + hi with per-operand strides.

struct GemmP {
  const float* X; const float* W; const float* bias; const float* res; float* Y;
  int M, N, K;
  int lda, ldb, ldy;
  int Hdiv;
  long long sxb, swb, syb;
  long long sxh, swh, syh;
  float scale;
};

template<int ACT_IN, int ACT_OUT, int HAS_BIAS, int HAS_RES, int BTRANS, int NTX>
__global__ void __launch_bounds__(256, 2) gemm_wmma(GemmP p) {
  constexpr int BM = 128, BN = 64, BK = 32;
  constexpr int LSH = 40;   // f16 staging row stride in halves (80B, 16B aligned)
  constexpr int LSF = 36;   // f32 async row stride in floats (144B, 16B aligned)
  constexpr bool UA = (HAVE_ASYNC != 0) && (BTRANS == 0) && (ACT_IN == 0);
  constexpr unsigned SBYTES = UA ? (unsigned)(2u * (BM * LSF + BN * LSF) * 4u)
                                 : (unsigned)((BM * LSH + BN * LSH) * 2u);
  __shared__ __align__(16) char smem[SBYTES];

  const int tid  = threadIdx.x;
  const int lane = tid & 31;
  const int wave = tid >> 5;
  const int wm   = wave & 3;   // 4 waves along M
  const int wn   = wave >> 2;  // 2 waves along N
  const int half = lane >> 4;
  const int l16  = lane & 15;

  const int bz = blockIdx.z;
  const int bi = bz / p.Hdiv, hi = bz % p.Hdiv;
  const float* Xb = p.X + (long long)bi * p.sxb + (long long)hi * p.sxh;
  const float* Wb = p.W + (long long)bi * p.swb + (long long)hi * p.swh;
  float*       Yb = p.Y + (long long)bi * p.syb + (long long)hi * p.syh;

  const int m_blk = blockIdx.x * BM;
  const int n_blk = blockIdx.y * BN;

  v8f acc[2][2];
  #pragma unroll
  for (int mi = 0; mi < 2; ++mi)
    #pragma unroll
    for (int ni = 0; ni < 2; ++ni)
      #pragma unroll
      for (int e = 0; e < 8; ++e) acc[mi][ni][e] = 0.f;

#if HAVE_ASYNC
  if constexpr (UA) {
    // ================= async global->LDS staging (f32 ping-pong) ==========
    float* Asf = (float*)smem;                              // [2][BM*LSF]
    float* Bsf = (float*)(smem + 2u * BM * LSF * 4u);       // [2][BN*LSF]

    auto asyncTile = [&](int k0, int buf) {
      float* Ad = Asf + buf * (BM * LSF);
      float* Bd = Bsf + buf * (BN * LSF);
      #pragma unroll
      for (int j = 0; j < 4; ++j) {               // A: 128x32 = 1024 float4
        int e = j * 256 + tid;
        int r = e >> 3, kq = (e & 7) * 4;
        int gm = min(m_blk + r, p.M - 1);         // clamp; OOB rows discarded
        long long off = (long long)gm * p.lda + (k0 + kq);
        __builtin_amdgcn_global_load_async_to_lds_b128(
            (__attribute__((address_space(1))) i32x4*)(Xb + off),
            (__attribute__((address_space(3))) i32x4*)(Ad + r * LSF + kq), 0, 0);
      }
      #pragma unroll
      for (int j = 0; j < 2; ++j) {               // B: 64x32 = 512 float4
        int e = j * 256 + tid;
        int c = e >> 3, kq = (e & 7) * 4;
        int gn = min(n_blk + c, p.N - 1);         // clamp; OOB cols discarded
        long long off = (long long)gn * p.ldb + (k0 + kq);
        __builtin_amdgcn_global_load_async_to_lds_b128(
            (__attribute__((address_space(1))) i32x4*)(Wb + off),
            (__attribute__((address_space(3))) i32x4*)(Bd + c * LSF + kq), 0, 0);
      }
    };

    const int nk = p.K / BK;                      // exact: K%32==0 on this path
    asyncTile(0, 0);
    for (int it = 0; it < nk; ++it) {
      const int buf = it & 1;
      const bool nxt = (it + 1) < nk;
      if (nxt) asyncTile((it + 1) * BK, buf ^ 1);
      // 6 async ops/wave/tile, in-order completion: <=6 outstanding ==> tile
      // 'it' has fully landed in LDS.
      if (nxt) __builtin_amdgcn_s_wait_asynccnt(6);
      else     __builtin_amdgcn_s_wait_asynccnt(0);
      __syncthreads();

      const float* Acur = Asf + buf * (BM * LSF);
      const float* Bcur = Bsf + buf * (BN * LSF);
      v16h afrag[2], bfrag[2];
      #pragma unroll
      for (int mi = 0; mi < 2; ++mi) {
        const float* base = Acur + (wm * 32 + mi * 16 + l16) * LSF;
        v4f c0 = *(const v4f*)(base + half * 16);
        v4f c1 = *(const v4f*)(base + half * 16 + 4);
        v4f c2 = *(const v4f*)(base + half * 16 + 8);
        v4f c3 = *(const v4f*)(base + half * 16 + 12);
        #pragma unroll
        for (int e = 0; e < 4; ++e) {
          afrag[mi][e]      = (_Float16)c0[e];
          afrag[mi][e + 4]  = (_Float16)c1[e];
          afrag[mi][e + 8]  = (_Float16)c2[e];
          afrag[mi][e + 12] = (_Float16)c3[e];
        }
      }
      #pragma unroll
      for (int ni = 0; ni < 2; ++ni) {
        const float* base = Bcur + (wn * 32 + ni * 16 + l16) * LSF + half * 16;
        v4f c0 = *(const v4f*)(base);
        v4f c1 = *(const v4f*)(base + 4);
        v4f c2 = *(const v4f*)(base + 8);
        v4f c3 = *(const v4f*)(base + 12);
        #pragma unroll
        for (int e = 0; e < 4; ++e) {
          bfrag[ni][e]      = (_Float16)c0[e];
          bfrag[ni][e + 4]  = (_Float16)c1[e];
          bfrag[ni][e + 8]  = (_Float16)c2[e];
          bfrag[ni][e + 12] = (_Float16)c3[e];
        }
      }
      #pragma unroll
      for (int mi = 0; mi < 2; ++mi)
        #pragma unroll
        for (int ni = 0; ni < 2; ++ni)
          acc[mi][ni] = __builtin_amdgcn_wmma_f32_16x16x32_f16(
              false, afrag[mi], false, bfrag[ni], (short)0, acc[mi][ni], false, false);
      __syncthreads();   // protect buf before the next iteration's prefetch overwrites it
    }
  } else
#endif
  {
    // ============== register-double-buffered f16 staging path ==============
    _Float16* As = (_Float16*)smem;
    _Float16* Bs = (_Float16*)(smem + BM * LSH * 2u);
    v4f  aR[4];
    v4f  bR[2];
    float bRs[8];

    auto loadA = [&](int k0) {
      #pragma unroll
      for (int j = 0; j < 4; ++j) {
        int e = j * 256 + tid;
        int r = e >> 3, kq = (e & 7) * 4;
        int gm = m_blk + r, gk = k0 + kq;
        int ok = (gm < p.M) & (gk < p.K);
        long long off = ok ? ((long long)gm * p.lda + gk) : 0;
        v4f v;
        if (NTX) v = __builtin_nontemporal_load((const v4f*)(Xb + off));
        else     v = *(const v4f*)(Xb + off);
        v *= (ok ? 1.f : 0.f);
        aR[j] = v;
      }
    };
    auto loadB = [&](int k0) {
      if (BTRANS == 0) {
        #pragma unroll
        for (int j = 0; j < 2; ++j) {
          int e = j * 256 + tid;
          int c = e >> 3, kq = (e & 7) * 4;
          int gn = n_blk + c, gk = k0 + kq;
          int ok = (gn < p.N) & (gk < p.K);
          long long off = ok ? ((long long)gn * p.ldb + gk) : 0;
          v4f v = *(const v4f*)(Wb + off);
          v *= (ok ? 1.f : 0.f);
          bR[j] = v;
        }
      } else {
        #pragma unroll
        for (int j = 0; j < 8; ++j) {
          int e = j * 256 + tid;
          int c = e >> 5, kk = e & 31;
          int gn = n_blk + c, gk = k0 + kk;
          int ok = (gn < p.N) & (gk < p.K);
          long long off = ok ? ((long long)gk * p.ldb + gn) : 0;
          bRs[j] = Wb[off] * (ok ? 1.f : 0.f);
        }
      }
    };
    auto storeA = [&]() {
      #pragma unroll
      for (int j = 0; j < 4; ++j) {
        int e = j * 256 + tid;
        int r = e >> 3, kq = (e & 7) * 4;
        v4f v = aR[j];
        if (ACT_IN == 1) {
          #pragma unroll
          for (int c = 0; c < 4; ++c) v[c] = v[c] / (1.f + __expf(-v[c]));
        }
        v4h h;
        #pragma unroll
        for (int c = 0; c < 4; ++c) h[c] = (_Float16)v[c];
        *(v4h*)&As[r * LSH + kq] = h;
      }
    };
    auto storeB = [&]() {
      if (BTRANS == 0) {
        #pragma unroll
        for (int j = 0; j < 2; ++j) {
          int e = j * 256 + tid;
          int c = e >> 3, kq = (e & 7) * 4;
          v4f v = bR[j];
          v4h h;
          #pragma unroll
          for (int cc = 0; cc < 4; ++cc) h[cc] = (_Float16)v[cc];
          *(v4h*)&Bs[c * LSH + kq] = h;
        }
      } else {
        #pragma unroll
        for (int j = 0; j < 8; ++j) {
          int e = j * 256 + tid;
          int c = e >> 5, kk = e & 31;
          Bs[c * LSH + kk] = (_Float16)bRs[j];
        }
      }
    };

    loadA(0); loadB(0);
    for (int k0 = 0; k0 < p.K; k0 += BK) {
      storeA(); storeB();
      __syncthreads();
      if (k0 + BK < p.K) { loadA(k0 + BK); loadB(k0 + BK); }

      v16h afrag[2], bfrag[2];
      #pragma unroll
      for (int mi = 0; mi < 2; ++mi) {
        const _Float16* base = &As[(wm * 32 + mi * 16 + l16) * LSH];
        v8h lo = *(const v8h*)(base + half * 16);
        v8h hh = *(const v8h*)(base + half * 16 + 8);
        #pragma unroll
        for (int e = 0; e < 8; ++e) { afrag[mi][e] = lo[e]; afrag[mi][e + 8] = hh[e]; }
      }
      #pragma unroll
      for (int ni = 0; ni < 2; ++ni) {
        const _Float16* base = &Bs[(wn * 32 + ni * 16 + l16) * LSH + half * 16];
        v8h lo = *(const v8h*)(base);
        v8h hh = *(const v8h*)(base + 8);
        #pragma unroll
        for (int e = 0; e < 8; ++e) { bfrag[ni][e] = lo[e]; bfrag[ni][e + 8] = hh[e]; }
      }
      #pragma unroll
      for (int mi = 0; mi < 2; ++mi)
        #pragma unroll
        for (int ni = 0; ni < 2; ++ni)
          acc[mi][ni] = __builtin_amdgcn_wmma_f32_16x16x32_f16(
              false, afrag[mi], false, bfrag[ni], (short)0, acc[mi][ni], false, false);
      __syncthreads();
    }
  }

  // ---- epilogue: C layout = VGPR i : row i + 8*half, col l16 ----
  const bool full = (m_blk + BM <= p.M) && (n_blk + BN <= p.N);
  #pragma unroll
  for (int mi = 0; mi < 2; ++mi) {
    #pragma unroll
    for (int ni = 0; ni < 2; ++ni) {
      #pragma unroll
      for (int i = 0; i < 8; ++i) {
        int grow = m_blk + wm * 32 + mi * 16 + i + half * 8;
        int gcol = n_blk + wn * 32 + ni * 16 + l16;
        if (full || (grow < p.M && gcol < p.N)) {
          float v = acc[mi][ni][i] * p.scale;
          if (HAS_BIAS) v += p.bias[gcol];
          if (HAS_RES)  v += p.res[(long long)grow * p.ldy + gcol];
          if (ACT_OUT == 1) v = fmaxf(v, 0.f);
          Yb[(long long)grow * p.ldy + gcol] = v;
        }
      }
    }
  }
}

// ---------------------------- elementwise add ------------------------------
__global__ void add_k(const float* __restrict__ a, const float* __restrict__ b,
                      float* __restrict__ o, int n4) {   // n4 = n/4
  int i = blockIdx.x * 256 + threadIdx.x;
  if (i < n4) {
    v4f va = *(const v4f*)(a + 4 * (long long)i);
    v4f vb = *(const v4f*)(b + 4 * (long long)i);
    *(v4f*)(o + 4 * (long long)i) = va + vb;
  }
}

// ----------------------- softmax over rows of length 300 -------------------
__global__ void softmax300_k(float* __restrict__ s, int rows) {
  int row = blockIdx.x * 8 + (threadIdx.x >> 5);
  if (row >= rows) return;
  int lane = threadIdx.x & 31;
  float* x = s + (long long)row * 300;
  float v[10], vmax = -1e30f;
  #pragma unroll
  for (int j = 0; j < 10; ++j) {
    int c = lane + j * 32;
    float t = x[c < 300 ? c : 299];
    v[j] = (c < 300) ? t : -1e30f;
    vmax = fmaxf(vmax, v[j]);
  }
  #pragma unroll
  for (int off = 16; off; off >>= 1) vmax = fmaxf(vmax, __shfl_xor(vmax, off, 32));
  float sum = 0.f;
  #pragma unroll
  for (int j = 0; j < 10; ++j) { v[j] = __expf(v[j] - vmax); sum += v[j]; }
  #pragma unroll
  for (int off = 16; off; off >>= 1) sum += __shfl_xor(sum, off, 32);
  float r = 1.f / sum;
  #pragma unroll
  for (int j = 0; j < 10; ++j) {
    int c = lane + j * 32;
    if (c < 300) x[c] = v[j] * r;
  }
}

// ------------------- softmax over groups of 16 (attn weights) --------------
__global__ void softmax16_k(float* __restrict__ x, int groups) {
  int g = blockIdx.x * 256 + threadIdx.x;
  if (g >= groups) return;
  float* p = x + (long long)g * 16;
  v4f u[4];
  #pragma unroll
  for (int i = 0; i < 4; ++i) u[i] = *(const v4f*)(p + 4 * i);
  float m = -1e30f;
  #pragma unroll
  for (int i = 0; i < 4; ++i)
    #pragma unroll
    for (int c = 0; c < 4; ++c) m = fmaxf(m, u[i][c]);
  float s = 0.f;
  #pragma unroll
  for (int i = 0; i < 4; ++i)
    #pragma unroll
    for (int c = 0; c < 4; ++c) { u[i][c] = __expf(u[i][c] - m); s += u[i][c]; }
  float r = 1.f / s;
  #pragma unroll
  for (int i = 0; i < 4; ++i) *(v4f*)(p + 4 * i) = u[i] * r;
}

// ------------------------- layernorm (D=256), wave/row ---------------------
template<int MOD>
__global__ void layernorm_k(const float* __restrict__ in, const float* __restrict__ g,
                            const float* __restrict__ b, const float* __restrict__ ss,
                            float* __restrict__ out, int rows) {
  int row = blockIdx.x * 8 + (threadIdx.x >> 5);
  if (row >= rows) return;
  int lane = threadIdx.x & 31;
  const float* x = in + (long long)row * 256 + lane * 8;
  v4f u0 = *(const v4f*)(x);
  v4f u1 = *(const v4f*)(x + 4);
  float s = 0.f;
  #pragma unroll
  for (int c = 0; c < 4; ++c) s += u0[c] + u1[c];
  #pragma unroll
  for (int off = 16; off; off >>= 1) s += __shfl_xor(s, off, 32);
  float mean = s * (1.f / 256.f);
  float var = 0.f;
  #pragma unroll
  for (int c = 0; c < 4; ++c) {
    float d0 = u0[c] - mean, d1 = u1[c] - mean;
    var += d0 * d0 + d1 * d1;
  }
  #pragma unroll
  for (int off = 16; off; off >>= 1) var += __shfl_xor(var, off, 32);
  float rstd = rsqrtf(var * (1.f / 256.f) + 1e-5f);

  v4f g0 = *(const v4f*)(g + lane * 8), g1 = *(const v4f*)(g + lane * 8 + 4);
  v4f b0 = *(const v4f*)(b + lane * 8), b1 = *(const v4f*)(b + lane * 8 + 4);
  v4f y0, y1;
  #pragma unroll
  for (int c = 0; c < 4; ++c) {
    y0[c] = (u0[c] - mean) * rstd * g0[c] + b0[c];
    y1[c] = (u1[c] - mean) * rstd * g1[c] + b1[c];
  }
  if (MOD) {
    const float* sp = ss + (long long)row * 512 + lane * 8;
    v4f sc0 = *(const v4f*)(sp),        sc1 = *(const v4f*)(sp + 4);
    v4f sh0 = *(const v4f*)(sp + 256),  sh1 = *(const v4f*)(sp + 260);
    #pragma unroll
    for (int c = 0; c < 4; ++c) {
      y0[c] = y0[c] * (sc0[c] + 1.f) + sh0[c];
      y1[c] = y1[c] * (sc1[c] + 1.f) + sh1[c];
    }
  }
  float* o = out + (long long)row * 256 + lane * 8;
  *(v4f*)(o) = y0;
  *(v4f*)(o + 4) = y1;
}

// ---------------------- deformable sampling core ---------------------------
__global__ void deform_k(const float* __restrict__ val,   // (B, S=7680, 256)
                         const float* __restrict__ offs,  // (B*LQ, 128)
                         const float* __restrict__ aw,    // (B*LQ, 128) softmaxed
                         const float* __restrict__ ref,   // (B*LQ, 4)
                         float* __restrict__ ca,          // (B*LQ, 256)
                         int total_waves) {
  int wid = blockIdx.x * 8 + (threadIdx.x >> 5);
  if (wid >= total_waves) return;
  int lane = threadIdx.x & 31;
  int h  = wid & 7;
  int bq = wid >> 3;
  int b  = bq / 300;
  const int T[4]  = {4096, 2048, 1024, 512};
  const int S0[4] = {0, 4096, 6144, 7168};
  const float* vb = val + (long long)b * 7680 * 256 + h * 32 + lane;
  const float* of = offs + (long long)bq * 128 + h * 16;
  const float* wv = aw   + (long long)bq * 128 + h * 16;
  const float* rf = ref  + (long long)bq * 4;
  float acc = 0.f;
  #pragma unroll
  for (int l = 0; l < 4; ++l) {
    float r = rf[l];
    float lenf = (float)T[l];
    const float* vl = vb + (long long)S0[l] * 256;
    #pragma unroll
    for (int pp = 0; pp < 4; ++pp) {
      float o  = of[l * 4 + pp];
      float w  = wv[l * 4 + pp];
      float xx = (r + o / lenf) * lenf - 0.5f;
      float x0 = floorf(xx);
      float f  = xx - x0;
      int  i0  = (int)x0;
      int  i1  = i0 + 1;
      int  i0c = min(max(i0, 0), T[l] - 1);
      int  i1c = min(max(i1, 0), T[l] - 1);
      float m0 = (i0 >= 0 && i0 < T[l]) ? 1.f : 0.f;
      float m1 = (i1 >= 0 && i1 < T[l]) ? 1.f : 0.f;
      float g0 = vl[(long long)i0c * 256] * m0;
      float g1 = vl[(long long)i1c * 256] * m1;
      acc += w * (g0 * (1.f - f) + g1 * f);
    }
  }
  ca[(long long)bq * 256 + h * 32 + lane] = acc;
}

// ------------------------------ host side ----------------------------------
static inline GemmP gp(const float* X, const float* W, const float* bias,
                       const float* res, float* Y, int M, int N, int K,
                       int lda, int ldb, int ldy, float scale = 1.f,
                       int Hdiv = 1,
                       long long sxb = 0, long long swb = 0, long long syb = 0,
                       long long sxh = 0, long long swh = 0, long long syh = 0) {
  GemmP p;
  p.X = X; p.W = W; p.bias = bias; p.res = res; p.Y = Y;
  p.M = M; p.N = N; p.K = K; p.lda = lda; p.ldb = ldb; p.ldy = ldy;
  p.Hdiv = Hdiv; p.sxb = sxb; p.swb = swb; p.syb = syb;
  p.sxh = sxh; p.swh = swh; p.syh = syh; p.scale = scale;
  return p;
}
static inline dim3 ggrid(int M, int N, int batches) {
  return dim3((unsigned)((M + 127) / 128), (unsigned)((N + 63) / 64), (unsigned)batches);
}

extern "C" void kernel_launch(void* const* d_in, const int* in_sizes, int n_in,
                              void* d_out, int out_size, void* d_ws, size_t ws_size,
                              hipStream_t stream) {
  (void)in_sizes; (void)n_in; (void)out_size; (void)ws_size;
  const float* tgt   = (const float*)d_in[0];
  const float* qpos  = (const float*)d_in[1];
  const float* temb  = (const float*)d_in[2];
  const float* refp  = (const float*)d_in[3];
  const float* src   = (const float*)d_in[4];
  const float* in_w  = (const float*)d_in[5];
  const float* in_b  = (const float*)d_in[6];
  const float* op_w  = (const float*)d_in[7];
  const float* op_b  = (const float*)d_in[8];
  const float* n1g   = (const float*)d_in[9];
  const float* n1b   = (const float*)d_in[10];
  const float* n2g   = (const float*)d_in[11];
  const float* n2b   = (const float*)d_in[12];
  const float* n3g   = (const float*)d_in[13];
  const float* n3b   = (const float*)d_in[14];
  const float* vp_w  = (const float*)d_in[15];
  const float* vp_b  = (const float*)d_in[16];
  const float* so_w  = (const float*)d_in[17];
  const float* so_b  = (const float*)d_in[18];
  const float* awp_w = (const float*)d_in[19];
  const float* awp_b = (const float*)d_in[20];
  const float* o2_w  = (const float*)d_in[21];
  const float* o2_b  = (const float*)d_in[22];
  const float* l1_w  = (const float*)d_in[23];
  const float* l1_b  = (const float*)d_in[24];
  const float* l2_w  = (const float*)d_in[25];
  const float* l2_b  = (const float*)d_in[26];
  const float* tw    = (const float*)d_in[27];
  const float* tb    = (const float*)d_in[28];

  float* ws  = (float*)d_ws;
  float* out = (float*)d_out;

  const int BQ  = 32 * 300;        // 9600 rows
  const int BQD = BQ * 256;        // 2,457,600
  // transient region (later reused by the 252MB 'value' tensor)
  const long long oQ   = 0;
  const long long oQK  = oQ   + BQD;                 // (9600,512)
  const long long oV   = oQK  + (long long)BQ * 512;
  const long long oS   = oV   + BQD;                 // scores (256,300,300)
  const long long oSA  = oS   + 256LL * 300 * 300;
  const long long oO   = oSA  + BQD;                 // tgt + self_attn
  const long long oVal = 0;
  // persistent region (above end of 'value' = 62,914,560 floats)
  const long long PB     = 62914560LL;
  const long long oX2    = PB;
  const long long oQuery = oX2    + BQD;
  const long long oOffs  = oQuery + BQD;
  const long long oAw    = oOffs  + (long long)BQ * 128;
  const long long oCa    = oAw    + (long long)BQ * 128;
  const long long oCa2   = oCa    + BQD;
  const long long oSS    = oCa2   + BQD;
  const long long oX1    = oSS    + (long long)BQ * 512;
  const long long oH     = oX1    + BQD;
  const long long oH2    = oH     + (long long)BQ * 1024;

  // 1. q = tgt + query_pos
  add_k<<<dim3(BQD / 4 / 256), 256, 0, stream>>>(tgt, qpos, ws + oQ, BQD / 4);

  // 2. [Q|K] = q @ in_proj_w[0:512].T + b
  gemm_wmma<0,0,1,0,0,0><<<ggrid(BQ, 512, 1), 256, 0, stream>>>(
      gp(ws + oQ, in_w, in_b, nullptr, ws + oQK, BQ, 512, 256, 256, 256, 512));

  // 3. V = tgt @ in_proj_w[512:768].T + b
  gemm_wmma<0,0,1,0,0,0><<<ggrid(BQ, 256, 1), 256, 0, stream>>>(
      gp(tgt, in_w + 512LL * 256, in_b + 512, nullptr, ws + oV, BQ, 256, 256, 256, 256, 256));

  // 4. scores = Q @ K.T / sqrt(32), batched over (b,h)=256
  gemm_wmma<0,0,0,0,0,0><<<ggrid(300, 300, 256), 256, 0, stream>>>(
      gp(ws + oQK, ws + oQK + 256, nullptr, nullptr, ws + oS,
         300, 300, 32, 512, 512, 300, 0.17677669529663687f, 8,
         300LL * 512, 300LL * 512, 8LL * 300 * 300,
         32, 32, 300LL * 300));

  // 5. softmax over rows of 300
  softmax300_k<<<dim3((256 * 300) / 8), 256, 0, stream>>>(ws + oS, 256 * 300);

  // 6. sa = P @ V  (B in (K,N) layout, K=300 tail -> register path)
  gemm_wmma<0,0,0,0,1,0><<<ggrid(300, 32, 256), 256, 0, stream>>>(
      gp(ws + oS, ws + oV, nullptr, nullptr, ws + oSA,
         300, 32, 300, 300, 256, 256, 1.f, 8,
         8LL * 300 * 300, 300LL * 256, 300LL * 256,
         300LL * 300, 32, 32));

  // 7. o = sa @ out_proj_w.T + b + tgt
  gemm_wmma<0,0,1,1,0,0><<<ggrid(BQ, 256, 1), 256, 0, stream>>>(
      gp(ws + oSA, op_w, op_b, tgt, ws + oO, BQ, 256, 256, 256, 256, 256));

  // 8. x2 = LN(o)
  layernorm_k<0><<<dim3(BQ / 8), 256, 0, stream>>>(ws + oO, n2g, n2b, nullptr, ws + oX2, BQ);

  // 9. query = x2 + query_pos
  add_k<<<dim3(BQD / 4 / 256), 256, 0, stream>>>(ws + oX2, qpos, ws + oQuery, BQD / 4);

  // 10. value = src @ value_proj_w.T + b  (async staging of 252MB stream)
  gemm_wmma<0,0,1,0,0,1><<<ggrid(32 * 7680, 256, 1), 256, 0, stream>>>(
      gp(src, vp_w, vp_b, nullptr, ws + oVal, 32 * 7680, 256, 256, 256, 256, 256));

  // 11. sampling offsets = query @ so_w.T + so_b
  gemm_wmma<0,0,1,0,0,0><<<ggrid(BQ, 128, 1), 256, 0, stream>>>(
      gp(ws + oQuery, so_w, so_b, nullptr, ws + oOffs, BQ, 128, 256, 256, 256, 128));

  // 12. attn weights (pre-softmax) = query @ aw_w.T + aw_b
  gemm_wmma<0,0,1,0,0,0><<<ggrid(BQ, 128, 1), 256, 0, stream>>>(
      gp(ws + oQuery, awp_w, awp_b, nullptr, ws + oAw, BQ, 128, 256, 256, 256, 128));

  // 13. softmax over L*P=16 per (b,q,h), in place
  softmax16_k<<<dim3((BQ * 8 + 255) / 256), 256, 0, stream>>>(ws + oAw, BQ * 8);

  // 14. deformable gather -> ca  (one wave per (b,q,h))
  deform_k<<<dim3(BQ), 256, 0, stream>>>(ws + oVal, ws + oOffs, ws + oAw, refp,
                                         ws + oCa, BQ * 8);

  // 15. ca2 = ca @ out2_w.T + b + x2
  gemm_wmma<0,0,1,1,0,0><<<ggrid(BQ, 256, 1), 256, 0, stream>>>(
      gp(ws + oCa, o2_w, o2_b, ws + oX2, ws + oCa2, BQ, 256, 256, 256, 256, 256));

  // 16. ss = silu(time_embed) @ time_w.T + time_b  (silu -> register path, NT)
  gemm_wmma<1,0,1,0,0,1><<<ggrid(BQ, 512, 1), 256, 0, stream>>>(
      gp(temb, tw, tb, nullptr, ws + oSS, BQ, 512, 1024, 1024, 1024, 512));

  // 17. x1 = LN(ca2)*(scale+1)+shift
  layernorm_k<1><<<dim3(BQ / 8), 256, 0, stream>>>(ws + oCa2, n1g, n1b, ws + oSS, ws + oX1, BQ);

  // 18. h = relu(x1 @ lin1_w.T + b)
  gemm_wmma<0,1,1,0,0,0><<<ggrid(BQ, 1024, 1), 256, 0, stream>>>(
      gp(ws + oX1, l1_w, l1_b, nullptr, ws + oH, BQ, 1024, 256, 256, 256, 1024));

  // 19. h2 = h @ lin2_w.T + b + x1
  gemm_wmma<0,0,1,1,0,0><<<ggrid(BQ, 256, 1), 256, 0, stream>>>(
      gp(ws + oH, l2_w, l2_b, ws + oX1, ws + oH2, BQ, 256, 1024, 1024, 1024, 256));

  // 20. out = LN(h2)
  layernorm_k<0><<<dim3(BQ / 8), 256, 0, stream>>>(ws + oH2, n3g, n3b, nullptr, out, BQ);
}
